// Seq2Seq_18468359373221
// MI455X (gfx1250) — matrix operand order repaired and checked by reading, not compiled, not run on hardware
//
#include <hip/hip_runtime.h>
#include <stdint.h>

// ---------------------------------------------------------------------------
// CDNA5 (gfx1250) wave32 WMMA types and helpers
// ---------------------------------------------------------------------------
typedef __attribute__((ext_vector_type(16))) __bf16 bf16x16;
typedef __attribute__((ext_vector_type(8)))  float  f32x8;

struct U8x32 { uint32_t v[8]; };   // 32 bytes == bf16x16

__device__ __forceinline__ uint16_t f2bf(float f) {
  uint32_t u = __builtin_bit_cast(uint32_t, f);
  u += 0x7FFFu + ((u >> 16) & 1u);            // round-to-nearest-even
  return (uint16_t)(u >> 16);
}
__device__ __forceinline__ float bf2f(uint16_t h) {
  uint32_t u = ((uint32_t)h) << 16;
  return __builtin_bit_cast(float, u);
}
__device__ __forceinline__ uint32_t pack2(float a, float b) {
  return (uint32_t)f2bf(a) | ((uint32_t)f2bf(b) << 16);
}
__device__ __forceinline__ bf16x16 frag_combine(uint4 a, uint4 b) {
  U8x32 u;
  u.v[0] = a.x; u.v[1] = a.y; u.v[2] = a.z; u.v[3] = a.w;
  u.v[4] = b.x; u.v[5] = b.y; u.v[6] = b.z; u.v[7] = b.w;
  return __builtin_bit_cast(bf16x16, u);
}
// A-fragment: 8 halves at p[0..7], 8 halves at p[16..23] (split chunks)
__device__ __forceinline__ bf16x16 load_frag_split(const uint16_t* __restrict__ p) {
  return frag_combine(*(const uint4*)(p), *(const uint4*)(p + 16));
}
// B-fragment: 8 halves at p[0..7], 8 halves at p[16..23] (split chunks)
__device__ __forceinline__ bf16x16 load_frag_contig(const uint16_t* __restrict__ p) {
  return frag_combine(*(const uint4*)(p), *(const uint4*)(p + 16));
}
__device__ __forceinline__ f32x8 wmma_bf16(bf16x16 a, bf16x16 b, f32x8 c) {
  return __builtin_amdgcn_wmma_f32_16x16x32_bf16(false, a, false, b, (short)0, c,
                                                 false, false);
}
__device__ __forceinline__ float sigm(float x) { return 1.0f / (1.0f + expf(-x)); }

// async global -> LDS copy of 16 bytes (ASYNCcnt-tracked, per-lane addresses)
__device__ __forceinline__ void async_g2l_b128(uint32_t lds_off, const void* g) {
  asm volatile("global_load_async_to_lds_b128 %0, %1, off"
               :: "v"(lds_off), "v"((uint64_t)(uintptr_t)g) : "memory");
}
__device__ __forceinline__ void wait_asynccnt0() {
  asm volatile("s_wait_asynccnt 0x0" ::: "memory");
}

#define SEQ 48
#define BSZ 16
#define HID 1024
#define EMB 512
#define NMIX 15
#define VOC 32000
#define TDEC 16
#define PADTOK 0
#define STARTTOK 1

// ---------------------------------------------------------------------------
// Vectorized f32 -> bf16 conversion (8 elements/thread); n % 8 == 0
// ---------------------------------------------------------------------------
__global__ void k_cvt8(const float* __restrict__ in, uint16_t* __restrict__ out,
                       int n8) {
  const int idx = blockIdx.x * blockDim.x + threadIdx.x;
  if (idx >= n8) return;
  const float4 f0 = ((const float4*)in)[2 * idx];
  const float4 f1 = ((const float4*)in)[2 * idx + 1];
  uint4 o;
  o.x = pack2(f0.x, f0.y);
  o.y = pack2(f0.z, f0.w);
  o.z = pack2(f1.x, f1.y);
  o.w = pack2(f1.z, f1.w);
  ((uint4*)out)[idx] = o;
}

// ---------------------------------------------------------------------------
// WMMA GEMM:  C(M,N) = A_bf16(M,K) @ W_bf16(N,K)^T + bias
// Block = 32 A-rows x 512 columns; wave = 2 M-tiles x 4 N-tiles (8 WMMAs per
// k-step, B fragments software-pipelined one k-step ahead).
// USELDS=1 (K<=1024): A tile async-copied to swizzled LDS once per block via
// global_load_async_to_lds_b128 + s_wait_asynccnt; USELDS=0: global A loads.
// M % 16 == 0, K % 32 == 0, N arbitrary (row-clamped loads + store guards).
// ---------------------------------------------------------------------------
__device__ __forceinline__ int aidx(int m, int k) {   // swizzled LDS index
  return (m << 10) + ((((k >> 3) ^ (m & 7)) << 3) | (k & 7));
}

template <int USELDS>
__global__ void __launch_bounds__(256) k_gemm(
    const uint16_t* __restrict__ A, int lda,
    const uint16_t* __restrict__ W, int ldw,
    const float* __restrict__ bias,
    float* __restrict__ C, int ldc,
    uint16_t* __restrict__ Cb, int ldcb,
    int M, int N, int K) {
  __shared__ uint16_t sA[32 * 1024];               // 64 KB (USELDS path)
  const int lane = threadIdx.x & 31;
  const int wid  = threadIdx.x >> 5;
  const int lm   = lane & 15;
  const int kA = (lane >> 4) << 3;   // A chunk offset within 32-k step
  const int kB = (lane >> 4) << 3;   // B chunk offset within 32-k step
  const int tilesM  = M >> 4;
  const int tilesN  = (N + 15) >> 4;
  const int groupsM = (tilesM + 1) >> 1;
  const int superN  = (tilesN + 31) >> 5;
  const int nTasks  = groupsM * superN;

  for (int task = blockIdx.x; task < nTasks; task += gridDim.x) {
    const int mg = task / superN;
    const int sg = task % superN;

    if (USELDS) {
      __syncthreads();                             // previous task done with sA
      const int cpr = K >> 3;                      // 8-half chunks per row
      for (int c = threadIdx.x; c < 32 * cpr; c += 256) {
        const int m = c / cpr, kc = c % cpr;
        int row = (mg << 5) + m;
        if (row >= M) row = M - 1;
        const uint16_t* g = A + (size_t)row * lda + (kc << 3);
        const uint32_t ld =
            (uint32_t)(uintptr_t)&sA[(m << 10) + (((kc ^ (m & 7)) << 3))];
        async_g2l_b128(ld, g);
      }
      wait_asynccnt0();
      __syncthreads();
    }

    const int tn0 = (sg << 5) + (wid << 2);        // wave's first n-tile
    const uint16_t* wp[4];
#pragma unroll
    for (int u = 0; u < 4; ++u) {
      int n = ((tn0 + u) << 4) + lm;
      if (n >= N) n = N - 1;                       // clamp: garbage guarded later
      wp[u] = W + (size_t)n * ldw + kB;
    }
    const int tm0 = mg << 1;
    const uint16_t* ap0;
    const uint16_t* ap1;
    if (!USELDS) {
      int r0 = (tm0 << 4) + lm;          if (r0 >= M) r0 = M - 1;
      int r1 = ((tm0 + 1) << 4) + lm;    if (r1 >= M) r1 = M - 1;
      ap0 = A + (size_t)r0 * lda + kA;
      ap1 = A + (size_t)r1 * lda + kA;
    }

    f32x8 acc[2][4];
#pragma unroll
    for (int s = 0; s < 2; ++s)
#pragma unroll
      for (int u = 0; u < 4; ++u)
        acc[s][u] = (f32x8){0.f,0.f,0.f,0.f,0.f,0.f,0.f,0.f};

    bf16x16 bc[4];
#pragma unroll
    for (int u = 0; u < 4; ++u) bc[u] = load_frag_contig(wp[u]);

    for (int kb = 0; kb < K; kb += 32) {
      const int kn = kb + 32;
      bf16x16 bn[4];
      if (kn < K) {                                // prefetch next B k-step
#pragma unroll
        for (int u = 0; u < 4; ++u) bn[u] = load_frag_contig(wp[u] + kn);
      }
      bf16x16 a0, a1;
      if (USELDS) {
        const int k0 = kb + kA;
        a0 = frag_combine(*(const uint4*)&sA[aidx(lm, k0)],
                          *(const uint4*)&sA[aidx(lm, k0 + 16)]);
        a1 = frag_combine(*(const uint4*)&sA[aidx(16 + lm, k0)],
                          *(const uint4*)&sA[aidx(16 + lm, k0 + 16)]);
      } else {
        a0 = load_frag_split(ap0 + kb);
        a1 = load_frag_split(ap1 + kb);
      }
#pragma unroll
      for (int u = 0; u < 4; ++u) {
        acc[0][u] = wmma_bf16(a0, bc[u], acc[0][u]);
        acc[1][u] = wmma_bf16(a1, bc[u], acc[1][u]);
      }
      if (kn < K) {
#pragma unroll
        for (int u = 0; u < 4; ++u) bc[u] = bn[u];
      }
    }

#pragma unroll
    for (int s = 0; s < 2; ++s) {
      const int tmu = tm0 + s;
      if (tmu >= tilesM) continue;                 // duplicated M-tile guard
      const int mrow0 = (tmu << 4) + ((lane >> 4) << 3);
#pragma unroll
      for (int u = 0; u < 4; ++u) {
        const int ncol = ((tn0 + u) << 4) + lm;
        if (ncol < N) {
          const float bv = bias ? bias[ncol] : 0.f;
#pragma unroll
          for (int r = 0; r < 8; ++r) {
            const float v = acc[s][u][r] + bv;
            C[(size_t)(mrow0 + r) * ldc + ncol] = v;
            if (Cb) Cb[(size_t)(mrow0 + r) * ldcb + ncol] = f2bf(v);
          }
        }
      }
    }
  }
}

// ---------------------------------------------------------------------------
// Sequential GRU over S steps (single workgroup, 16 waves).
// gi (S*B, 3H) precomputed (includes bih). gh = h@Whh^T on WMMA, Whh bf16
// software-pipelined one k-step ahead; h resident in LDS as bf16 with
// 8-half-chunk XOR swizzle; one A fragment feeds 3 WMMAs (r,z,n).
// ---------------------------------------------------------------------------
__device__ __forceinline__ int hidx(int m, int k) {
  return m * HID + ((((k >> 3) ^ (m & 7)) << 3) | (k & 7));
}

__global__ void __launch_bounds__(512) k_gru_seq(
    const float* __restrict__ gi,
    const uint16_t* __restrict__ Whh,   // (3H, H) bf16
    const float* __restrict__ bhh,      // (3H)
    uint16_t* __restrict__ ybf, int ldyb, int ycol,
    float* __restrict__ yf32, int ldyf,
    float* __restrict__ hT,
    int S, int reverse) {
  __shared__ uint16_t shb[BSZ * HID];        // 32 KB, h in bf16
  const int lane = threadIdx.x & 31;
  const int wid  = threadIdx.x >> 5;         // 0..15
  const int kA   = (lane >> 4) << 3;
  const int kB   = (lane >> 4) << 3;
  const int mloc = lane & 15;
  const int m0   = (lane >> 4) << 3;

  for (int i = threadIdx.x; i < BSZ * HID; i += 512) shb[i] = 0;
  __syncthreads();

  for (int tt = 0; tt < S; ++tt) {
    const int t = reverse ? (S - 1 - tt) : tt;
    float hnew[4][8];
#pragma unroll 1
    for (int ti = 0; ti < 4; ++ti) {
      const int jc = ((((wid << 2) + ti) << 4)) + (lane & 15);  // 64 hidden tiles
      const uint16_t* wR = Whh + (size_t)jc * HID + kB;
      const uint16_t* wZ = wR + (size_t)HID * HID;
      const uint16_t* wN = wZ + (size_t)HID * HID;
      f32x8 aR = {0.f,0.f,0.f,0.f,0.f,0.f,0.f,0.f};
      f32x8 aZ = aR, aN = aR;
      bf16x16 bR = load_frag_contig(wR);
      bf16x16 bZ = load_frag_contig(wZ);
      bf16x16 bN = load_frag_contig(wN);
      for (int kb = 0; kb < HID; kb += 32) {
        const int kn = kb + 32;
        bf16x16 nR, nZ, nN;
        if (kn < HID) {                       // prefetch next k-step of Whh
          nR = load_frag_contig(wR + kn);
          nZ = load_frag_contig(wZ + kn);
          nN = load_frag_contig(wN + kn);
        }
        const int k0 = kb + kA;
        const bf16x16 af = frag_combine(*(const uint4*)&shb[hidx(mloc, k0)],
                                        *(const uint4*)&shb[hidx(mloc, k0 + 16)]);
        aR = wmma_bf16(af, bR, aR);
        aZ = wmma_bf16(af, bZ, aZ);
        aN = wmma_bf16(af, bN, aN);
        if (kn < HID) { bR = nR; bZ = nZ; bN = nN; }
      }
      const float br = bhh[jc], bz = bhh[HID + jc], bn = bhh[2 * HID + jc];
#pragma unroll
      for (int r = 0; r < 8; ++r) {
        const int m = m0 + r;
        const size_t grow = (size_t)(t * BSZ + m) * (3 * HID);
        const float rg = sigm(gi[grow + jc] + aR[r] + br);
        const float zg = sigm(gi[grow + HID + jc] + aZ[r] + bz);
        const float ng = tanhf(gi[grow + 2 * HID + jc] + rg * (aN[r] + bn));
        const float ho = bf2f(shb[hidx(m, jc)]);
        hnew[ti][r] = (1.f - zg) * ng + zg * ho;
      }
    }
    __syncthreads();   // everyone done reading old h
#pragma unroll
    for (int ti = 0; ti < 4; ++ti) {
      const int jc = ((((wid << 2) + ti) << 4)) + (lane & 15);
#pragma unroll
      for (int r = 0; r < 8; ++r) {
        const int m = m0 + r;
        const float v = hnew[ti][r];
        shb[hidx(m, jc)] = f2bf(v);
        const size_t row = (size_t)(t * BSZ + m);
        ybf[row * ldyb + ycol + jc] = f2bf(v);
        yf32[row * ldyf + jc] = v;
      }
    }
    __syncthreads();
  }
  for (int i = threadIdx.x; i < BSZ * HID; i += 512) {
    const int m = i >> 10, k = i & (HID - 1);
    hT[i] = bf2f(shb[hidx(m, k)]);
  }
}

// ---------------------------------------------------------------------------
// Small elementwise / reduction kernels
// ---------------------------------------------------------------------------
__global__ void k_embed_enc(const int* __restrict__ inputs,
                            const float* __restrict__ emb,
                            uint16_t* __restrict__ out) {
  const int idx = blockIdx.x * blockDim.x + threadIdx.x;   // 768*512
  if (idx >= SEQ * BSZ * EMB) return;
  const int row = idx >> 9, e = idx & (EMB - 1);
  const int tok = inputs[row];
  const float v = (tok == PADTOK) ? 0.f : emb[(size_t)tok * EMB + e];
  out[idx] = f2bf(v);
}

__global__ void k_embed_dec(const int* __restrict__ tok,
                            const float* __restrict__ emb,
                            uint16_t* __restrict__ out, int t) {
  const int idx = blockIdx.x * blockDim.x + threadIdx.x;   // 16*512
  if (idx >= BSZ * EMB) return;
  const int b = idx >> 9, e = idx & (EMB - 1);
  const int tk = (t == 0) ? STARTTOK : tok[b];
  out[idx] = f2bf(emb[(size_t)tk * EMB + e]);
}

__global__ void k_dec_init(const float* __restrict__ hf, const float* __restrict__ hb,
                           const float* __restrict__ h1, const float* __restrict__ h2,
                           float* __restrict__ sA, float* __restrict__ sB, float* __restrict__ sC,
                           uint16_t* __restrict__ sAb, uint16_t* __restrict__ sBb,
                           uint16_t* __restrict__ sCb) {
  const int idx = blockIdx.x * blockDim.x + threadIdx.x;   // 3*16384
  if (idx >= 3 * BSZ * HID) return;
  const int sec = idx / (BSZ * HID), i = idx % (BSZ * HID);
  float v;
  if (sec == 0)      { v = hf[i] + hb[i]; sA[i] = v; sAb[i] = f2bf(v); }
  else if (sec == 1) { v = h1[i];         sB[i] = v; sBb[i] = f2bf(v); }
  else               { v = h2[i];         sC[i] = v; sCb[i] = f2bf(v); }
}

__global__ void k_gate(const float* __restrict__ gi, const float* __restrict__ gh,
                       float* __restrict__ h, uint16_t* __restrict__ hb) {
  const int idx = blockIdx.x * blockDim.x + threadIdx.x;   // 16384
  if (idx >= BSZ * HID) return;
  const int b = idx >> 10, j = idx & (HID - 1);
  const size_t r0 = (size_t)b * 3 * HID;
  const float rg = sigm(gi[r0 + j] + gh[r0 + j]);
  const float zg = sigm(gi[r0 + HID + j] + gh[r0 + HID + j]);
  const float ng = tanhf(gi[r0 + 2 * HID + j] + rg * gh[r0 + 2 * HID + j]);
  const float v = (1.f - zg) * ng + zg * h[idx];
  h[idx] = v;
  hb[idx] = f2bf(v);
}

__global__ void __launch_bounds__(512) k_attn(
    const float* __restrict__ q, const float* __restrict__ enc_out,
    const int* __restrict__ inputs, const float* __restrict__ c,
    uint16_t* __restrict__ Acomb) {
  __shared__ float sw[BSZ][SEQ];
  for (int idx = threadIdx.x; idx < BSZ * SEQ; idx += 512) {
    const int b = idx / SEQ, s = idx % SEQ;
    float sc;
    if (inputs[s * BSZ + b] == PADTOK) {
      sc = -__builtin_inff();
    } else {
      sc = 0.f;
      const float* qp = q + (size_t)b * HID;
      const float* ep = enc_out + (size_t)(s * BSZ + b) * HID;
      for (int hh = 0; hh < HID; ++hh) sc += qp[hh] * ep[hh];
    }
    sw[b][s] = sc;
  }
  __syncthreads();
  if (threadIdx.x < BSZ) {
    const int b = threadIdx.x;
    float mx = -__builtin_inff();
    for (int s = 0; s < SEQ; ++s) mx = fmaxf(mx, sw[b][s]);
    float sum = 0.f;
    for (int s = 0; s < SEQ; ++s) {
      const float e = (sw[b][s] == -__builtin_inff()) ? 0.f : expf(sw[b][s] - mx);
      sw[b][s] = e;
      sum += e;
    }
    const float inv = 1.f / sum;
    for (int s = 0; s < SEQ; ++s) sw[b][s] *= inv;
  }
  __syncthreads();
  for (int idx = threadIdx.x; idx < BSZ * HID; idx += 512) {
    const int b = idx >> 10, hh = idx & (HID - 1);
    float ctx = 0.f;
    for (int s = 0; s < SEQ; ++s)
      ctx += sw[b][s] * enc_out[(size_t)(s * BSZ + b) * HID + hh];
    Acomb[(size_t)b * 2 * HID + HID + hh] = f2bf(ctx);
    Acomb[(size_t)b * 2 * HID + hh] = f2bf(c[idx]);
  }
}

__global__ void k_tanh_bf16(const float* __restrict__ in, uint16_t* __restrict__ out, int n) {
  const int idx = blockIdx.x * blockDim.x + threadIdx.x;
  if (idx < n) out[idx] = f2bf(tanhf(in[idx]));
}

__global__ void k_prior(const float* __restrict__ pl, float* __restrict__ pr) {
  const int b = threadIdx.x;
  if (b >= BSZ) return;
  float mx = -__builtin_inff();
  for (int k = 0; k < NMIX; ++k) mx = fmaxf(mx, pl[b * 16 + k]);
  float sum = 0.f, e[NMIX];
  for (int k = 0; k < NMIX; ++k) { e[k] = expf(pl[b * 16 + k] - mx); sum += e[k]; }
  const float inv = 1.f / sum;
  for (int k = 0; k < NMIX; ++k) pr[b * NMIX + k] = e[k] * inv;
}

__global__ void __launch_bounds__(256) k_rowstat(const float* __restrict__ logits,
                                                 float* __restrict__ rmax,
                                                 float* __restrict__ rsum) {
  __shared__ float red[256];
  const int row = blockIdx.x;
  const float* p = logits + (size_t)row * VOC;
  float mx = -__builtin_inff();
  for (int v = threadIdx.x; v < VOC; v += 256) mx = fmaxf(mx, p[v]);
  red[threadIdx.x] = mx;
  __syncthreads();
  for (int s = 128; s > 0; s >>= 1) {
    if (threadIdx.x < s) red[threadIdx.x] = fmaxf(red[threadIdx.x], red[threadIdx.x + s]);
    __syncthreads();
  }
  mx = red[0];
  __syncthreads();
  float sum = 0.f;
  for (int v = threadIdx.x; v < VOC; v += 256) sum += expf(p[v] - mx);
  red[threadIdx.x] = sum;
  __syncthreads();
  for (int s = 128; s > 0; s >>= 1) {
    if (threadIdx.x < s) red[threadIdx.x] += red[threadIdx.x + s];
    __syncthreads();
  }
  if (threadIdx.x == 0) { rmax[row] = mx; rsum[row] = red[0]; }
}

__global__ void __launch_bounds__(256) k_mix(
    const float* __restrict__ logits, const float* __restrict__ rmax,
    const float* __restrict__ rsum, const float* __restrict__ pr,
    float* __restrict__ scores, float* __restrict__ pval, int* __restrict__ pidx) {
  __shared__ float bv[256];
  __shared__ int bi[256];
  const int b = blockIdx.y, chunk = blockIdx.x;
  const int v = chunk * 256 + threadIdx.x;   // 125*256 == 32000
  float p = 0.f;
#pragma unroll
  for (int k = 0; k < NMIX; ++k) {
    const int r = b * NMIX + k;
    p += pr[r] * expf(logits[(size_t)r * VOC + v] - rmax[r]) / rsum[r];
  }
  const float lp = logf(p + 1e-8f);
  scores[(size_t)b * VOC + v] = lp;
  bv[threadIdx.x] = lp;
  bi[threadIdx.x] = v;
  __syncthreads();
  for (int s = 128; s > 0; s >>= 1) {
    if (threadIdx.x < s) {
      const float ov = bv[threadIdx.x + s];
      const int oi = bi[threadIdx.x + s];
      if (ov > bv[threadIdx.x] || (ov == bv[threadIdx.x] && oi < bi[threadIdx.x])) {
        bv[threadIdx.x] = ov; bi[threadIdx.x] = oi;
      }
    }
    __syncthreads();
  }
  if (threadIdx.x == 0) { pval[b * 125 + chunk] = bv[0]; pidx[b * 125 + chunk] = bi[0]; }
}

__global__ void __launch_bounds__(128) k_argmax_final(
    const float* __restrict__ pval, const int* __restrict__ pidx,
    int* __restrict__ tok, float* __restrict__ preds) {
  __shared__ float bv[128];
  __shared__ int bi[128];
  const int b = blockIdx.x;
  float best = -__builtin_inff();
  int besti = 0x7fffffff;
  for (int i = threadIdx.x; i < 125; i += 128) {
    const float v = pval[b * 125 + i];
    const int id = pidx[b * 125 + i];
    if (v > best || (v == best && id < besti)) { best = v; besti = id; }
  }
  bv[threadIdx.x] = best;
  bi[threadIdx.x] = besti;
  __syncthreads();
  for (int s = 64; s > 0; s >>= 1) {
    if (threadIdx.x < s) {
      const float ov = bv[threadIdx.x + s];
      const int oi = bi[threadIdx.x + s];
      if (ov > bv[threadIdx.x] || (ov == bv[threadIdx.x] && oi < bi[threadIdx.x])) {
        bv[threadIdx.x] = ov; bi[threadIdx.x] = oi;
      }
    }
    __syncthreads();
  }
  if (threadIdx.x == 0) { tok[b] = bi[0]; preds[b] = (float)bi[0]; }
}

// ---------------------------------------------------------------------------
// Host driver
// ---------------------------------------------------------------------------
extern "C" void kernel_launch(void* const* d_in, const int* in_sizes, int n_in,
                              void* d_out, int out_size, void* d_ws, size_t ws_size,
                              hipStream_t stream) {
  (void)in_sizes; (void)n_in; (void)out_size; (void)ws_size;

  const int*   inputs      = (const int*)  d_in[0];
  const float* enc_emb     = (const float*)d_in[2];
  const float* e0f_Wih     = (const float*)d_in[3];
  const float* e0f_Whh     = (const float*)d_in[4];
  const float* e0f_bih     = (const float*)d_in[5];
  const float* e0f_bhh     = (const float*)d_in[6];
  const float* e0b_Wih     = (const float*)d_in[7];
  const float* e0b_Whh     = (const float*)d_in[8];
  const float* e0b_bih     = (const float*)d_in[9];
  const float* e0b_bhh     = (const float*)d_in[10];
  const float* merge_W     = (const float*)d_in[11];
  const float* e1_Wih      = (const float*)d_in[12];
  const float* e1_Whh      = (const float*)d_in[13];
  const float* e1_bih      = (const float*)d_in[14];
  const float* e1_bhh      = (const float*)d_in[15];
  const float* e2_Wih      = (const float*)d_in[16];
  const float* e2_Whh      = (const float*)d_in[17];
  const float* e2_bih      = (const float*)d_in[18];
  const float* e2_bhh      = (const float*)d_in[19];
  const float* dec_emb     = (const float*)d_in[20];
  const float* d0_Wih      = (const float*)d_in[21];
  const float* d0_Whh      = (const float*)d_in[22];
  const float* d0_bih      = (const float*)d_in[23];
  const float* d0_bhh      = (const float*)d_in[24];
  const float* d1_Wih      = (const float*)d_in[25];
  const float* d1_Whh      = (const float*)d_in[26];
  const float* d1_bih      = (const float*)d_in[27];
  const float* d1_bhh      = (const float*)d_in[28];
  const float* d2_Wih      = (const float*)d_in[29];
  const float* d2_Whh      = (const float*)d_in[30];
  const float* d2_bih      = (const float*)d_in[31];
  const float* d2_bhh      = (const float*)d_in[32];
  const float* attn_proj_W = (const float*)d_in[33];
  const float* attn_comb_W = (const float*)d_in[34];
  const float* prior_W     = (const float*)d_in[35];
  const float* latent_W    = (const float*)d_in[36];
  const float* latent_b    = (const float*)d_in[37];
  const float* out_bias    = (const float*)d_in[38];

  float* out = (float*)d_out;

  // bump allocator over workspace
  char* wpb_ = (char*)d_ws;
  auto alloc = [&](size_t bytes) -> void* {
    void* p = (void*)wpb_;
    wpb_ += (bytes + 255) & ~(size_t)255;
    return p;
  };
  // f32 scratch
  float*    logits  = (float*)   alloc((size_t)BSZ * NMIX * VOC * 4);
  float*    gibuf   = (float*)   alloc((size_t)SEQ * BSZ * 3 * HID * 4);
  float*    scrap   = (float*)   alloc((size_t)SEQ * BSZ * 2 * HID * 4);
  float*    enc_out = (float*)   alloc((size_t)SEQ * BSZ * HID * 4);
  float*    hf      = (float*)   alloc((size_t)BSZ * HID * 4);
  float*    hb      = (float*)   alloc((size_t)BSZ * HID * 4);
  float*    h1      = (float*)   alloc((size_t)BSZ * HID * 4);
  float*    h2      = (float*)   alloc((size_t)BSZ * HID * 4);
  float*    sA      = (float*)   alloc((size_t)BSZ * HID * 4);
  float*    sB      = (float*)   alloc((size_t)BSZ * HID * 4);
  float*    sC      = (float*)   alloc((size_t)BSZ * HID * 4);
  float*    gid     = (float*)   alloc((size_t)BSZ * 3 * HID * 4);
  float*    ghd     = (float*)   alloc((size_t)BSZ * 3 * HID * 4);
  float*    qb      = (float*)   alloc((size_t)BSZ * HID * 4);
  float*    comb    = (float*)   alloc((size_t)BSZ * HID * 4);
  float*    latp    = (float*)   alloc((size_t)BSZ * NMIX * EMB * 4);
  float*    priorlg = (float*)   alloc((size_t)BSZ * 16 * 4);
  float*    pr      = (float*)   alloc((size_t)BSZ * NMIX * 4);
  float*    rmax    = (float*)   alloc((size_t)BSZ * NMIX * 4);
  float*    rsum    = (float*)   alloc((size_t)BSZ * NMIX * 4);
  float*    pval    = (float*)   alloc((size_t)BSZ * 125 * 4);
  int*      pidx    = (int*)     alloc((size_t)BSZ * 125 * 4);
  int*      tok     = (int*)     alloc((size_t)BSZ * 4);
  // bf16 activation staging
  uint16_t* Aemb    = (uint16_t*)alloc((size_t)SEQ * BSZ * EMB * 2);
  uint16_t* Amerge  = (uint16_t*)alloc((size_t)SEQ * BSZ * 2 * HID * 2);
  uint16_t* A1      = (uint16_t*)alloc((size_t)SEQ * BSZ * HID * 2);
  uint16_t* A2      = (uint16_t*)alloc((size_t)SEQ * BSZ * HID * 2);
  uint16_t* sAb     = (uint16_t*)alloc((size_t)BSZ * HID * 2);
  uint16_t* sBb     = (uint16_t*)alloc((size_t)BSZ * HID * 2);
  uint16_t* sCb     = (uint16_t*)alloc((size_t)BSZ * HID * 2);
  uint16_t* xb      = (uint16_t*)alloc((size_t)BSZ * EMB * 2);
  uint16_t* Acomb   = (uint16_t*)alloc((size_t)BSZ * 2 * HID * 2);
  uint16_t* combb   = (uint16_t*)alloc((size_t)BSZ * HID * 2);
  uint16_t* latb    = (uint16_t*)alloc((size_t)BSZ * NMIX * EMB * 2);
  // bf16 weight staging
  auto walloc = [&](size_t n) { return (uint16_t*)alloc(n * 2); };
  uint16_t* w_e0f_ih = walloc((size_t)3 * HID * EMB);
  uint16_t* w_e0f_hh = walloc((size_t)3 * HID * HID);
  uint16_t* w_e0b_ih = walloc((size_t)3 * HID * EMB);
  uint16_t* w_e0b_hh = walloc((size_t)3 * HID * HID);
  uint16_t* w_merge  = walloc((size_t)HID * 2 * HID);
  uint16_t* w_e1_ih  = walloc((size_t)3 * HID * HID);
  uint16_t* w_e1_hh  = walloc((size_t)3 * HID * HID);
  uint16_t* w_e2_ih  = walloc((size_t)3 * HID * HID);
  uint16_t* w_e2_hh  = walloc((size_t)3 * HID * HID);
  uint16_t* w_d0_ih  = walloc((size_t)3 * HID * EMB);
  uint16_t* w_d0_hh  = walloc((size_t)3 * HID * HID);
  uint16_t* w_d1_ih  = walloc((size_t)3 * HID * HID);
  uint16_t* w_d1_hh  = walloc((size_t)3 * HID * HID);
  uint16_t* w_d2_ih  = walloc((size_t)3 * HID * HID);
  uint16_t* w_d2_hh  = walloc((size_t)3 * HID * HID);
  uint16_t* w_attnp  = walloc((size_t)HID * HID);
  uint16_t* w_attnc  = walloc((size_t)HID * 2 * HID);
  uint16_t* w_prior  = walloc((size_t)NMIX * HID);
  uint16_t* w_lat    = walloc((size_t)NMIX * EMB * HID);
  uint16_t* w_dec    = walloc((size_t)VOC * EMB);

  auto cvt = [&](const float* src, uint16_t* dst, size_t n) {
    const int n8 = (int)(n / 8);
    k_cvt8<<<(n8 + 255) / 256, 256, 0, stream>>>(src, dst, n8);
  };
  // stage all weights as bf16 (read-many, WMMA-native)
  cvt(e0f_Wih, w_e0f_ih, (size_t)3 * HID * EMB);
  cvt(e0f_Whh, w_e0f_hh, (size_t)3 * HID * HID);
  cvt(e0b_Wih, w_e0b_ih, (size_t)3 * HID * EMB);
  cvt(e0b_Whh, w_e0b_hh, (size_t)3 * HID * HID);
  cvt(merge_W, w_merge,  (size_t)HID * 2 * HID);
  cvt(e1_Wih,  w_e1_ih,  (size_t)3 * HID * HID);
  cvt(e1_Whh,  w_e1_hh,  (size_t)3 * HID * HID);
  cvt(e2_Wih,  w_e2_ih,  (size_t)3 * HID * HID);
  cvt(e2_Whh,  w_e2_hh,  (size_t)3 * HID * HID);
  cvt(d0_Wih,  w_d0_ih,  (size_t)3 * HID * EMB);
  cvt(d0_Whh,  w_d0_hh,  (size_t)3 * HID * HID);
  cvt(d1_Wih,  w_d1_ih,  (size_t)3 * HID * HID);
  cvt(d1_Whh,  w_d1_hh,  (size_t)3 * HID * HID);
  cvt(d2_Wih,  w_d2_ih,  (size_t)3 * HID * HID);
  cvt(d2_Whh,  w_d2_hh,  (size_t)3 * HID * HID);
  cvt(attn_proj_W, w_attnp, (size_t)HID * HID);
  cvt(attn_comb_W, w_attnc, (size_t)HID * 2 * HID);
  cvt(prior_W,     w_prior, (size_t)NMIX * HID);
  cvt(latent_W,    w_lat,   (size_t)NMIX * EMB * HID);
  cvt(dec_emb,     w_dec,   (size_t)VOC * EMB);

  auto gemm = [&](const uint16_t* A, int lda, const uint16_t* W, int ldw,
                  const float* bias, float* C, int ldc, uint16_t* Cb, int ldcb,
                  int M, int N, int K) {
    const int tilesM = M / 16, tilesN = (N + 15) / 16;
    const int groupsM = (tilesM + 1) / 2, superN = (tilesN + 31) / 32;
    int blocks = groupsM * superN;
    if (blocks > 4096) blocks = 4096;
    if (K <= 1024)
      k_gemm<1><<<blocks, 256, 0, stream>>>(A, lda, W, ldw, bias, C, ldc, Cb, ldcb,
                                            M, N, K);
    else
      k_gemm<0><<<blocks, 256, 0, stream>>>(A, lda, W, ldw, bias, C, ldc, Cb, ldcb,
                                            M, N, K);
  };

  // ---------------- Encoder ----------------
  k_embed_enc<<<(SEQ * BSZ * EMB) / 256, 256, 0, stream>>>(inputs, enc_emb, Aemb);

  // layer 0 forward
  gemm(Aemb, EMB, w_e0f_ih, EMB, e0f_bih, gibuf, 3 * HID, nullptr, 0,
       SEQ * BSZ, 3 * HID, EMB);
  k_gru_seq<<<1, 512, 0, stream>>>(gibuf, w_e0f_hh, e0f_bhh,
                                   Amerge, 2 * HID, 0, scrap, HID, hf, SEQ, 0);
  // layer 0 backward
  gemm(Aemb, EMB, w_e0b_ih, EMB, e0b_bih, gibuf, 3 * HID, nullptr, 0,
       SEQ * BSZ, 3 * HID, EMB);
  k_gru_seq<<<1, 512, 0, stream>>>(gibuf, w_e0b_hh, e0b_bhh,
                                   Amerge, 2 * HID, HID, scrap, HID, hb, SEQ, 1);
  // merge projection (K=2048 -> global-A path)
  gemm(Amerge, 2 * HID, w_merge, 2 * HID, nullptr, scrap, HID, A1, HID,
       SEQ * BSZ, HID, 2 * HID);
  // layer 1
  gemm(A1, HID, w_e1_ih, HID, e1_bih, gibuf, 3 * HID, nullptr, 0,
       SEQ * BSZ, 3 * HID, HID);
  k_gru_seq<<<1, 512, 0, stream>>>(gibuf, w_e1_hh, e1_bhh,
                                   A2, HID, 0, scrap, HID, h1, SEQ, 0);
  // layer 2 (f32 output kept as enc_out for attention)
  gemm(A2, HID, w_e2_ih, HID, e2_bih, gibuf, 3 * HID, nullptr, 0,
       SEQ * BSZ, 3 * HID, HID);
  k_gru_seq<<<1, 512, 0, stream>>>(gibuf, w_e2_hh, e2_bhh,
                                   A1, HID, 0, enc_out, HID, h2, SEQ, 0);

  k_dec_init<<<(3 * BSZ * HID) / 256, 256, 0, stream>>>(hf, hb, h1, h2,
                                                        sA, sB, sC, sAb, sBb, sCb);

  // ---------------- Decoder ----------------
  for (int t = 0; t < TDEC; ++t) {
    k_embed_dec<<<(BSZ * EMB + 255) / 256, 256, 0, stream>>>(tok, dec_emb, xb, t);

    // GRU cell d0
    gemm(xb, EMB, w_d0_ih, EMB, d0_bih, gid, 3 * HID, nullptr, 0, BSZ, 3 * HID, EMB);
    gemm(sAb, HID, w_d0_hh, HID, d0_bhh, ghd, 3 * HID, nullptr, 0, BSZ, 3 * HID, HID);
    k_gate<<<(BSZ * HID) / 256, 256, 0, stream>>>(gid, ghd, sA, sAb);
    // GRU cell d1
    gemm(sAb, HID, w_d1_ih, HID, d1_bih, gid, 3 * HID, nullptr, 0, BSZ, 3 * HID, HID);
    gemm(sBb, HID, w_d1_hh, HID, d1_bhh, ghd, 3 * HID, nullptr, 0, BSZ, 3 * HID, HID);
    k_gate<<<(BSZ * HID) / 256, 256, 0, stream>>>(gid, ghd, sB, sBb);
    // GRU cell d2
    gemm(sBb, HID, w_d2_ih, HID, d2_bih, gid, 3 * HID, nullptr, 0, BSZ, 3 * HID, HID);
    gemm(sCb, HID, w_d2_hh, HID, d2_bhh, ghd, 3 * HID, nullptr, 0, BSZ, 3 * HID, HID);
    k_gate<<<(BSZ * HID) / 256, 256, 0, stream>>>(gid, ghd, sC, sCb);

    // Luong attention
    gemm(sCb, HID, w_attnp, HID, nullptr, qb, HID, nullptr, 0, BSZ, HID, HID);
    k_attn<<<1, 512, 0, stream>>>(qb, enc_out, inputs, sC, Acomb);
    gemm(Acomb, 2 * HID, w_attnc, 2 * HID, nullptr, comb, HID, combb, HID,
         BSZ, HID, 2 * HID);

    // mixture-of-softmaxes head
    gemm(combb, HID, w_lat, HID, latent_b, latp, NMIX * EMB, nullptr, 0,
         BSZ, NMIX * EMB, HID);
    k_tanh_bf16<<<(BSZ * NMIX * EMB) / 256, 256, 0, stream>>>(latp, latb,
                                                              BSZ * NMIX * EMB);
    gemm(combb, HID, w_prior, HID, nullptr, priorlg, 16, nullptr, 0,
         BSZ, NMIX, HID);
    k_prior<<<1, 32, 0, stream>>>(priorlg, pr);

    // big GEMM: (240 x 512) @ dec_emb^T (512 x 32000) + out_bias
    gemm(latb, EMB, w_dec, EMB, out_bias, logits, VOC, nullptr, 0,
         BSZ * NMIX, VOC, EMB);

    k_rowstat<<<BSZ * NMIX, 256, 0, stream>>>(logits, rmax, rsum);
    k_mix<<<dim3(125, BSZ), 256, 0, stream>>>(logits, rmax, rsum, pr,
                                              out + (size_t)t * BSZ * VOC, pval, pidx);
    k_argmax_final<<<BSZ, 128, 0, stream>>>(pval, pidx, tok,
                                            out + (size_t)TDEC * BSZ * VOC + t * BSZ);
  }
}
